// LlamaMultiHeadAttention_47614007443848
// MI455X (gfx1250) — matrix operand order repaired and checked by reading, not compiled, not run on hardware
//
#include <hip/hip_runtime.h>
#include <stdint.h>

// ---------------------------------------------------------------------------
// Llama MHA block for MI455X (gfx1250, wave32, WMMA).
// bf16 storage / WMMA math, fp32 accumulation. Flash-attention style online
// softmax (never materializes the 512 MB fp32 score matrix). Workspace need: 96 MB.
// ---------------------------------------------------------------------------

#define BATCH 2
#define SEQ   2048
#define HID   2048
#define NH    16
#define HD    128
#define MTOK  (BATCH * SEQ)   // 4096 tokens

typedef __attribute__((ext_vector_type(16))) __bf16 v16bf;
typedef __attribute__((ext_vector_type(8)))  __bf16 v8bf;
typedef __attribute__((ext_vector_type(8)))  float  v8f;

union FragU { v16bf v; v8bf h[2]; };

__device__ __forceinline__ float bf2f(unsigned short u) {
  union { unsigned int i; float f; } w; w.i = ((unsigned int)u) << 16; return w.f;
}
__device__ __forceinline__ unsigned short f2bf(float f) {
  union { float f; unsigned int i; } w; w.f = f;
  unsigned int x = w.i;
  unsigned int r = x + 0x7fffu + ((x >> 16) & 1u);   // round-to-nearest-even
  return (unsigned short)(r >> 16);
}

__device__ __forceinline__ v8f wmma_bf16(v16bf a, v16bf b, v8f c) {
  // v_wmma_f32_16x16x32_bf16: (neg_a, A, neg_b, B, c_mod, C, reuse_a, reuse_b)
  return __builtin_amdgcn_wmma_f32_16x16x32_bf16(false, a, false, b, (short)0, c,
                                                 false, false);
}

// A-matrix 16x32 fragment from a row-major source. lane -> row (lane&15);
// K chunks per ISA layout: half0 = {0..7,16..23}, half1 = {8..15,24..31}.
__device__ __forceinline__ v16bf load_frag_a(const unsigned short* p0, int ld,
                                             int r, int half) {
  const unsigned short* p = p0 + (size_t)r * ld + half * 8;
  FragU u;
  u.h[0] = *reinterpret_cast<const v8bf*>(p);
  u.h[1] = *reinterpret_cast<const v8bf*>(p + 16);
  return u.v;
}

// B-matrix 32x16 fragment where the source stores B^T row-major
// (row index = n, col index = k). lane -> column n (lane&15);
// K chunks matching A layout: half0 = {0..7,16..23}, half1 = {8..15,24..31}.
__device__ __forceinline__ v16bf load_frag_b(const unsigned short* p0, int ld,
                                             int r, int half) {
  const unsigned short* p = p0 + (size_t)r * ld + half * 8;
  FragU u;
  u.h[0] = *reinterpret_cast<const v8bf*>(p);
  u.h[1] = *reinterpret_cast<const v8bf*>(p + 16);
  return u.v;
}

// ---------------------------------------------------------------------------
// fp32 -> bf16 convert
// ---------------------------------------------------------------------------
__global__ __launch_bounds__(256) void cvt_f32_bf16(const float* __restrict__ s,
                                                    unsigned short* __restrict__ d,
                                                    int n) {
  for (int i = blockIdx.x * blockDim.x + threadIdx.x; i < n;
       i += gridDim.x * blockDim.x)
    d[i] = f2bf(s[i]);
}

// ---------------------------------------------------------------------------
// C[M,N] = A[M,K] @ W[N,K]^T  (nn.Linear), bf16 inputs, f32 accumulate.
// 256 threads = 8 waves; block tile 128x128; wave tile 32x64 (8 accumulators).
// ---------------------------------------------------------------------------
template <bool F32OUT>
__global__ __launch_bounds__(256) void gemm_xwt(const unsigned short* __restrict__ A,
                                                const unsigned short* __restrict__ W,
                                                void* __restrict__ Cout,
                                                int M, int N, int K) {
  const int tid  = threadIdx.x;
  const int lane = tid & 31;
  const int wave = tid >> 5;
  const int r    = lane & 15;
  const int half = (lane >> 4) & 1;
  const int wm = wave >> 1, wn = wave & 1;
  const int row0 = blockIdx.y * 128 + wm * 32;
  const int col0 = blockIdx.x * 128 + wn * 64;

  v8f acc[2][4];
#pragma unroll
  for (int i = 0; i < 2; ++i)
#pragma unroll
    for (int t = 0; t < 4; ++t)
      acc[i][t] = (v8f){0.f, 0.f, 0.f, 0.f, 0.f, 0.f, 0.f, 0.f};

  for (int k0 = 0; k0 < K; k0 += 32) {
    v16bf a0 = load_frag_a(A + (size_t)row0 * K + k0, K, r, half);
    v16bf a1 = load_frag_a(A + (size_t)(row0 + 16) * K + k0, K, r, half);
#pragma unroll
    for (int t = 0; t < 4; ++t) {
      v16bf b = load_frag_b(W + (size_t)(col0 + t * 16) * K + k0, K, r, half);
      acc[0][t] = wmma_bf16(a0, b, acc[0][t]);
      acc[1][t] = wmma_bf16(a1, b, acc[1][t]);
    }
    if (k0 + 32 < K) {   // global_prefetch_b8 hints for next K-slice
      __builtin_prefetch(A + (size_t)(row0 + r) * K + k0 + 32, 0, 1);
      __builtin_prefetch(W + (size_t)(col0 + r) * K + k0 + 32, 0, 1);
    }
  }

#pragma unroll
  for (int i = 0; i < 2; ++i)
#pragma unroll
    for (int t = 0; t < 4; ++t)
#pragma unroll
      for (int j = 0; j < 8; ++j) {
        int row = row0 + i * 16 + half * 8 + j;   // C layout: m = vgpr + 8*half
        int col = col0 + t * 16 + r;              //           n = lane&15
        if (F32OUT)
          ((float*)Cout)[(size_t)row * N + col] = acc[i][t][j];
        else
          ((unsigned short*)Cout)[(size_t)row * N + col] = f2bf(acc[i][t][j]);
      }
}

// ---------------------------------------------------------------------------
// RoPE in-place on bf16 [MTOK, HID] viewed as (b, s, h, d) duplicate layout.
// ---------------------------------------------------------------------------
__global__ __launch_bounds__(256) void rope_inplace(unsigned short* __restrict__ x) {
  const int n = MTOK * NH * (HD / 2);
  for (int i = blockIdx.x * blockDim.x + threadIdx.x; i < n;
       i += gridDim.x * blockDim.x) {
    int ii  = i & 63;          // rotary pair index 0..63
    int t1  = i >> 6;
    int h   = t1 & (NH - 1);
    int tok = t1 >> 4;
    int pos = tok & (SEQ - 1);
    size_t base = (size_t)tok * HID + h * HD;
    // inv_freq = 10000^(-ii/64) ; ln(10000)/64 = 0.14391156831212787
    float inv = __expf(-(float)ii * 0.14391156831212787f);
    float th  = (float)pos * inv;
    float c = cosf(th), s = sinf(th);
    float x0 = bf2f(x[base + ii]);
    float x1 = bf2f(x[base + ii + 64]);
    x[base + ii]      = f2bf(x0 * c - x1 * s);
    x[base + ii + 64] = f2bf(x1 * c + x0 * s);
  }
}

// ---------------------------------------------------------------------------
// V [b,s,(h d)] -> VT [(b h), d, s]  (so P@V B-frags become contiguous rows)
// ---------------------------------------------------------------------------
__global__ __launch_bounds__(256) void transpose_v(const unsigned short* __restrict__ v,
                                                   unsigned short* __restrict__ vt) {
  const int n = MTOK * HID;
  for (int o = blockIdx.x * blockDim.x + threadIdx.x; o < n;
       o += gridDim.x * blockDim.x) {
    int s  = o & (SEQ - 1);
    int t  = o >> 11;
    int d  = t & (HD - 1);
    int bh = t >> 7;
    int h  = bh & (NH - 1);
    int b  = bh >> 4;
    vt[o] = v[((size_t)(b * SEQ + s)) * HID + h * HD + d];
  }
}

// ---------------------------------------------------------------------------
// Flash attention forward. Grid (32 bh, 16 q-blocks); 8 waves/block; each wave
// owns a 16-row Q tile and streams causal 32-key tiles.
// ---------------------------------------------------------------------------
__global__ __launch_bounds__(256) void attn_fwd(const unsigned short* __restrict__ Q,
                                                const unsigned short* __restrict__ Kb,
                                                const unsigned short* __restrict__ VT,
                                                unsigned short* __restrict__ O) {
  __shared__ unsigned short pbuf[8][16 * 32];   // per-wave 16x32 bf16 P tile

  const int tid  = threadIdx.x;
  const int lane = tid & 31;
  const int wave = tid >> 5;
  const int r    = lane & 15;
  const int half = (lane >> 4) & 1;
  const int bh = blockIdx.x;
  const int b  = bh >> 4;
  const int h  = bh & (NH - 1);
  const int q0 = (blockIdx.y * 8 + wave) * 16;
  const float SCALE = 0.08838834764831845f;     // 1/sqrt(128)
  const float NEG_BIG = -3.0e38f;

  // Q fragments for the whole d=128 (4 chunks of 32)
  size_t qbase = ((size_t)(b * SEQ + q0)) * HID + h * HD;
  v16bf qf[4];
#pragma unroll
  for (int c = 0; c < 4; ++c)
    qf[c] = load_frag_a(Q + qbase + 32 * c, HID, r, half);

  v8f acc[8];
#pragma unroll
  for (int nd = 0; nd < 8; ++nd)
    acc[nd] = (v8f){0.f, 0.f, 0.f, 0.f, 0.f, 0.f, 0.f, 0.f};
  float mrow[8], lrow[8];
#pragma unroll
  for (int j = 0; j < 8; ++j) { mrow[j] = NEG_BIG; lrow[j] = 0.f; }

  unsigned short* pb = &pbuf[wave][0];

  for (int kt = 0; kt < q0 + 16; kt += 32) {     // causal: kt <= q0 always
    // ---- scores: S[16q x 32k] = Q (16x128) @ K^T ------------------------
    v8f s0 = (v8f){0.f, 0.f, 0.f, 0.f, 0.f, 0.f, 0.f, 0.f};
    v8f s1 = (v8f){0.f, 0.f, 0.f, 0.f, 0.f, 0.f, 0.f, 0.f};
#pragma unroll
    for (int c = 0; c < 4; ++c) {
      v16bf k0f = load_frag_b(Kb + ((size_t)(b * SEQ + kt)) * HID + h * HD + 32 * c,
                              HID, r, half);
      v16bf k1f = load_frag_b(Kb + ((size_t)(b * SEQ + kt + 16)) * HID + h * HD + 32 * c,
                              HID, r, half);
      s0 = wmma_bf16(qf[c], k0f, s0);
      s1 = wmma_bf16(qf[c], k1f, s1);
    }

    // ---- online softmax (per row j+8*half, cols spread over 16 lanes) ---
    float pv[2][8];
#pragma unroll
    for (int j = 0; j < 8; ++j) {
      int qrow = q0 + half * 8 + j;
      float v0 = s0[j] * SCALE;
      float v1 = s1[j] * SCALE;
      if (kt + r > qrow)      v0 = NEG_BIG;   // causal mask
      if (kt + 16 + r > qrow) v1 = NEG_BIG;
      float mt = fmaxf(v0, v1);
      mt = fmaxf(mt, __shfl_xor(mt, 1, 32));  // masks <16 stay inside half
      mt = fmaxf(mt, __shfl_xor(mt, 2, 32));
      mt = fmaxf(mt, __shfl_xor(mt, 4, 32));
      mt = fmaxf(mt, __shfl_xor(mt, 8, 32));
      float mn = fmaxf(mrow[j], mt);
      float al = __expf(mrow[j] - mn);
      float p0 = __expf(v0 - mn);
      float p1 = __expf(v1 - mn);
      float rs = p0 + p1;
      rs += __shfl_xor(rs, 1, 32);
      rs += __shfl_xor(rs, 2, 32);
      rs += __shfl_xor(rs, 4, 32);
      rs += __shfl_xor(rs, 8, 32);
      lrow[j] = lrow[j] * al + rs;
      mrow[j] = mn;
#pragma unroll
      for (int nd = 0; nd < 8; ++nd) acc[nd][j] *= al;
      pv[0][j] = p0;
      pv[1][j] = p1;
    }

    // ---- stage P (C layout -> row-major LDS -> A layout) ----------------
#pragma unroll
    for (int t = 0; t < 2; ++t)
#pragma unroll
      for (int j = 0; j < 8; ++j)
        pb[(half * 8 + j) * 32 + t * 16 + r] = f2bf(pv[t][j]);
    asm volatile("s_wait_dscnt 0" ::: "memory");
    v16bf pf = load_frag_a(pb, 32, r, half);

    // ---- out += P (16x32) @ V (32x128) ----------------------------------
#pragma unroll
    for (int nd = 0; nd < 8; ++nd) {
      v16bf vf = load_frag_b(VT + ((size_t)(bh * HD + nd * 16)) * SEQ + kt,
                             SEQ, r, half);
      acc[nd] = wmma_bf16(pf, vf, acc[nd]);
    }
  }

  // ---- epilogue: normalize and store bf16 [b, s, (h d)] -----------------
#pragma unroll
  for (int nd = 0; nd < 8; ++nd)
#pragma unroll
    for (int j = 0; j < 8; ++j) {
      int row = q0 + half * 8 + j;
      float o = acc[nd][j] / lrow[j];
      O[((size_t)(b * SEQ + row)) * HID + h * HD + nd * 16 + r] = f2bf(o);
    }
}

// ---------------------------------------------------------------------------
// Host-side launch
// ---------------------------------------------------------------------------
extern "C" void kernel_launch(void* const* d_in, const int* in_sizes, int n_in,
                              void* d_out, int out_size, void* d_ws, size_t ws_size,
                              hipStream_t stream) {
  (void)in_sizes; (void)n_in; (void)out_size; (void)ws_size;

  const float* X  = (const float*)d_in[0];
  const float* Wq = (const float*)d_in[1];
  const float* Wk = (const float*)d_in[2];
  const float* Wv = (const float*)d_in[3];
  const float* Wo = (const float*)d_in[4];
  float* out = (float*)d_out;

  uint8_t* ws = (uint8_t*)d_ws;
  const size_t MB = 1024 * 1024;
  // Workspace (96 MB total), with lifetime-based aliasing:
  unsigned short* Xbf    = (unsigned short*)(ws + 0);        // 16 MB (dead after QKV gemms)
  unsigned short* attnbf = Xbf;                              // reuses Xbf
  unsigned short* Wqbf   = (unsigned short*)(ws + 16 * MB);  // 8 MB
  unsigned short* Wkbf   = (unsigned short*)(ws + 24 * MB);  // 8 MB
  unsigned short* Wvbf   = (unsigned short*)(ws + 32 * MB);  // 8 MB
  unsigned short* Wobf   = (unsigned short*)(ws + 40 * MB);  // 8 MB (live to end)
  unsigned short* VTb    = (unsigned short*)(ws + 16 * MB);  // 16 MB, reuses Wq/Wk after their gemms
  unsigned short* Qbf    = (unsigned short*)(ws + 48 * MB);  // 16 MB
  unsigned short* Kbf    = (unsigned short*)(ws + 64 * MB);  // 16 MB
  unsigned short* Vbf    = (unsigned short*)(ws + 80 * MB);  // 16 MB

  const int nAct = MTOK * HID;   // 8,388,608
  const int nW   = HID * HID;    // 4,194,304

  cvt_f32_bf16<<<2048, 256, 0, stream>>>(X,  Xbf,  nAct);
  cvt_f32_bf16<<<2048, 256, 0, stream>>>(Wq, Wqbf, nW);
  cvt_f32_bf16<<<2048, 256, 0, stream>>>(Wk, Wkbf, nW);
  cvt_f32_bf16<<<2048, 256, 0, stream>>>(Wv, Wvbf, nW);
  cvt_f32_bf16<<<2048, 256, 0, stream>>>(Wo, Wobf, nW);

  dim3 ggrid(HID / 128, MTOK / 128);   // (16, 32)
  gemm_xwt<false><<<ggrid, 256, 0, stream>>>(Xbf, Wqbf, Qbf, MTOK, HID, HID);
  gemm_xwt<false><<<ggrid, 256, 0, stream>>>(Xbf, Wkbf, Kbf, MTOK, HID, HID);
  gemm_xwt<false><<<ggrid, 256, 0, stream>>>(Xbf, Wvbf, Vbf, MTOK, HID, HID);

  rope_inplace<<<4096, 256, 0, stream>>>(Qbf);
  rope_inplace<<<4096, 256, 0, stream>>>(Kbf);

  transpose_v<<<4096, 256, 0, stream>>>(Vbf, VTb);

  attn_fwd<<<dim3(BATCH * NH, SEQ / 128), 256, 0, stream>>>(Qbf, Kbf, VTb, attnbf);

  gemm_xwt<true><<<ggrid, 256, 0, stream>>>(attnbf, Wobf, out, MTOK, HID, HID);
}
